// ConvLayer_36790689858048
// MI455X (gfx1250) — compile-verified, matrix operand order checked by reading, not run on hardware
//
#include <hip/hip_runtime.h>

typedef __attribute__((ext_vector_type(16))) _Float16 v16h;
typedef __attribute__((ext_vector_type(8)))  _Float16 v8h;
typedef __attribute__((ext_vector_type(8)))  float    v8f;
typedef __attribute__((ext_vector_type(4)))  unsigned int u32x4;
typedef __attribute__((ext_vector_type(8)))  int      i32x8;
typedef __attribute__((ext_vector_type(4)))  int      i32x4;

#if defined(__has_builtin)
#  if __has_builtin(__builtin_amdgcn_tensor_load_to_lds)
#    define HAVE_TDM 1
#  endif
#endif
#if __has_include(<hip/amd_detail/amd_gfx1250_TDM.h>)
#  define TDM_ARITY6 1
#endif

#define NB    2
#define NN    500
#define NV    64
#define NW    5
#define NG    80
#define NGP   96      // G padded to multiple of 32 for K-tiling
#define NO    80
#define NROWS (NB*NN) // 1000
#define TILE  16
#define EPSV  1e-5f
#define TWO_PI 6.28318530717958647692f

__global__ __launch_bounds__(256)
void geoconv_wmma_kernel(const float* __restrict__ feat,     // [ROWS,V,W]
                         const float* __restrict__ rho,      // [ROWS,V]
                         const float* __restrict__ theta,    // [ROWS,V]
                         const float* __restrict__ mask,     // [ROWS,V]
                         const float* __restrict__ mu_rho,   // [W,G]
                         const float* __restrict__ mu_theta, // [W,G]
                         const float* __restrict__ sig_rho,  // [W,G]
                         const float* __restrict__ sig_theta,// [W,G]
                         const float* __restrict__ Wc,       // [W,G,O]
                         const float* __restrict__ bc,       // [W,O]
                         const int*   __restrict__ nrot_p,   // scalar
                         float* __restrict__ out)            // [ROWS,W,O]
{
    __shared__ float srho[TILE][NV];
    __shared__ float sth [TILE][NV];
    __shared__ float sthr[TILE][NV];
    __shared__ float smsk[TILE][NV];
    __shared__ float sft [NW][TILE][NV];
    __shared__ float smur[NW*NG], smut[NW*NG], sisr[NW*NG], sist[NW*NG];
    __shared__ __align__(16) _Float16 sdesc[NW][TILE][NGP];
    __shared__ __align__(16) _Float16 sWt[NW][NO][NGP];   // f16 W_conv, K-contig per column
    __shared__ __align__(16) float    sOut[NW][TILE][NO]; // doubles as f32 TDM staging (6400 floats)

    const int tid   = threadIdx.x;
    const int lane  = tid & 31;
    const int wave  = tid >> 5;
    const int tile0 = blockIdx.x * TILE;
    const int nrot  = nrot_p[0];

    __builtin_prefetch(Wc, 0, 1);
    __builtin_prefetch(feat, 0, 1);

    // ---- Prologue: stage rotation-invariant data in LDS ----
    for (int i = tid; i < TILE*NV; i += 256) {
        int r = i / NV, v = i % NV;
        int row = tile0 + r;
        float rv = 0.f, tv = 0.f, mv = 0.f;
        if (row < NROWS) {
            rv = rho  [row*NV + v];
            tv = theta[row*NV + v];
            mv = mask [row*NV + v];
        }
        srho[r][v] = rv; sth[r][v] = tv; smsk[r][v] = mv;
    }
    for (int i = tid; i < NW*TILE*NV; i += 256) {
        int w = i / (TILE*NV);
        int rem = i % (TILE*NV);
        int r = rem / NV, v = rem % NV;
        int row = tile0 + r;
        sft[w][r][v] = (row < NROWS) ? feat[(row*NV + v)*NW + w] : 0.f;
    }
    for (int i = tid; i < NW*NG; i += 256) {
        smur[i] = mu_rho[i];
        smut[i] = mu_theta[i];
        float sr = sig_rho[i], st = sig_theta[i];
        sisr[i] = 1.f / (sr*sr + EPSV);
        sist[i] = 1.f / (st*st + EPSV);
    }
    // zero whole desc once (keeps K-pad [80,96) exactly zero for all rotations)
    for (int i = tid; i < NW*TILE*NGP; i += 256)
        ((_Float16*)sdesc)[i] = (_Float16)0.f;

    // ---- W_conv -> f16 LDS tile, K-contiguous per output column ----
#ifdef HAVE_TDM
    {
        float* stage = &sOut[0][0][0];                       // 6400-float staging area
        unsigned ldsaddr = (unsigned)(unsigned long long)(unsigned long long)(uintptr_t)(void*)stage;
        for (int w = 0; w < NW; ++w) {
            if (wave == 0) {
                unsigned long long ga =
                    (unsigned long long)(uintptr_t)(const void*)(Wc + (size_t)w * NG * NO);
                // D# group 0: count=1, lds_addr, global_addr, type=2 (ISA 8.3)
                u32x4 g0 = { 1u,
                             ldsaddr,
                             (unsigned)(ga & 0xFFFFFFFFu),
                             (unsigned)((ga >> 32) & 0x01FFFFFFu) | 0x80000000u };
                // D# group 1: data_size=4B; 1-D tile of 6400 elems; stride 6400 (ISA 8.4)
                i32x8 g1 = { (int)0x00020000,          // data_size=2 (4 bytes)
                             (int)(6400u << 16),       // tensor_dim0[15:0] in bits 63:48
                             (int)(1u << 16),          // tensor_dim0 hi=0, tensor_dim1=1
                             (int)(6400u << 16),       // tile_dim0=6400 in bits 127:112
                             0,                        // tile_dim1=0, tile_dim2=0
                             6400,                     // tensor_dim0_stride low
                             (int)(6400u << 16),       // stride hi=0, tensor_dim1_stride low
                             0 };
                i32x4 g2 = { 0, 0, 0, 0 };
                i32x4 g3 = { 0, 0, 0, 0 };
#ifdef TDM_ARITY6
                i32x8 g4 = { 0, 0, 0, 0, 0, 0, 0, 0 };
                __builtin_amdgcn_tensor_load_to_lds(g0, g1, g2, g3, g4, 0);
#else
                __builtin_amdgcn_tensor_load_to_lds(g0, g1, g2, g3, 0);
#endif
                __builtin_amdgcn_s_wait_tensorcnt((short)0);
            }
            __syncthreads();                                 // staging visible to all waves
            for (int i = tid; i < NO*NGP; i += 256) {
                int o = i / NGP, kp = i % NGP;
                sWt[w][o][kp] = (_Float16)((kp < NG) ? stage[kp*NO + o] : 0.f);
            }
            __syncthreads();                                 // staging free for next w
        }
    }
#else
    for (int i = tid; i < NW*NO*NGP; i += 256) {
        int w  = i / (NO*NGP);
        int r2 = i % (NO*NGP);
        int o  = r2 / NGP, kp = r2 % NGP;
        sWt[w][o][kp] = (_Float16)((kp < NG) ? Wc[(w*NG + kp)*NO + o] : 0.f);
    }
#endif

    // running-max accumulator (after staging use of this storage)
    for (int i = tid; i < NW*TILE*NO; i += 256)
        ((float*)sOut)[i] = -3.402823466e38f;

    const float dstep = TWO_PI / (float)nrot;

    for (int k = 0; k < nrot; ++k) {
        __syncthreads();   // previous Stage B done reading sdesc; sOut init visible

        // ---- rotate thetas: th = mod(theta + k*2pi/nrot, 2pi) ----
        float koff = (float)k * dstep;
        for (int i = tid; i < TILE*NV; i += 256) {
            float t = ((const float*)sth)[i] + koff;   // t in [0, 4pi)
            t = (t >= TWO_PI) ? t - TWO_PI : t;
            ((float*)sthr)[i] = t;
        }
        __syncthreads();

        // ---- Stage A: desc[w][r][g] = (sum_v gauss*feat) / (sum_v gauss + eps) ----
        for (int j = tid; j < NW*TILE*NG; j += 256) {
            int g = j % NG;
            int r = (j / NG) % TILE;
            int w = j / (NG*TILE);
            float mr = smur[w*NG + g], mt = smut[w*NG + g];
            float ir = sisr[w*NG + g], it = sist[w*NG + g];
            float num = 0.f, den = 0.f;
            #pragma unroll 8
            for (int v = 0; v < NV; ++v) {
                float dr = srho[r][v] - mr;
                float dt = sthr[r][v] - mt;
                float e  = smsk[r][v] * __expf(-(dr*dr*ir + dt*dt*it));
                num += e * sft[w][r][v];
                den += e;
            }
            sdesc[w][r][g] = (_Float16)(num / (den + EPSV));
        }
        __syncthreads();

        // ---- Stage B: out_tile[w] = max(out_tile[w], desc[w] @ Wc[w]) via WMMA ----
        // 25 jobs = 5 w x 5 o-tiles, distributed across 8 waves (wave-uniform).
        for (int job = wave; job < NW*(NO/16); job += 8) {
            int w  = job / (NO/16);
            int ot = job % (NO/16);
            int m  = lane & 15;                 // A row / D column-lane
            int n  = lane & 15;                 // B/D column
            int hk = (lane >> 4) << 3;          // K half select: 0 or 8
            v8f c = {};
            #pragma unroll
            for (int kt = 0; kt < 3; ++kt) {
                int kb = kt * 32;
                // A fragment: 16-bit A 16x32 layout (ISA 7.12.2)
                v16h a, b;
                v8h a0 = *(const v8h*)&sdesc[w][m][kb + hk];
                v8h a1 = *(const v8h*)&sdesc[w][m][kb + 16 + hk];
                // B fragment: same K striping, column-contiguous LDS tile
                v8h b0 = *(const v8h*)&sWt[w][ot*16 + n][kb + hk];
                v8h b1 = *(const v8h*)&sWt[w][ot*16 + n][kb + 16 + hk];
                #pragma unroll
                for (int i = 0; i < 8; ++i) {
                    a[i] = a0[i]; a[8+i] = a1[i];
                    b[i] = b0[i]; b[8+i] = b1[i];
                }
                c = __builtin_amdgcn_wmma_f32_16x16x32_f16(
                        /*neg_a=*/false, a, /*neg_b=*/false, b,
                        /*c_mod=*/(short)0, c, /*reuse_a=*/false, /*reuse_b=*/false);
            }
            // D layout: lane -> column n, VGPR r -> row r (+8 for upper lane half)
            int mb = (lane >> 4) << 3;
            #pragma unroll
            for (int r = 0; r < 8; ++r) {
                float old = sOut[w][mb + r][ot*16 + n];
                sOut[w][mb + r][ot*16 + n] = fmaxf(old, c[r]);
            }
        }
    }
    __syncthreads();

    // ---- Epilogue: add bias (constant over rotations, commutes with max) ----
    for (int i = tid; i < NW*TILE*NO; i += 256) {
        int o = i % NO;
        int r = (i / NO) % TILE;
        int w = i / (NO*TILE);
        int row = tile0 + r;
        if (row < NROWS)
            out[(row*NW + w)*NO + o] = sOut[w][r][o] + bc[w*NO + o];
    }
}

extern "C" void kernel_launch(void* const* d_in, const int* in_sizes, int n_in,
                              void* d_out, int out_size, void* d_ws, size_t ws_size,
                              hipStream_t stream) {
    const float* feat  = (const float*)d_in[0];   // input_feat [B,N,V,W]
    const float* rho   = (const float*)d_in[1];   // rho_coords [B,N,V]
    const float* theta = (const float*)d_in[2];   // theta_coords [B,N,V]
    const float* msk   = (const float*)d_in[3];   // mask [B,N,V,1]
    const float* mur   = (const float*)d_in[4];   // mu_rho [W,G]
    const float* mut   = (const float*)d_in[5];   // mu_theta [W,G]
    const float* sr    = (const float*)d_in[6];   // sigma_rho [W,G]
    const float* st    = (const float*)d_in[7];   // sigma_theta [W,G]
    const float* Wc    = (const float*)d_in[8];   // W_conv [W,G,O]
    const float* bc    = (const float*)d_in[9];   // b_conv [W,O]
    const int*   nrot  = (const int*)  d_in[10];  // n_rotations scalar
    float* out = (float*)d_out;                   // [B,N,W,O]

    dim3 grid((NROWS + TILE - 1) / TILE);         // 63 workgroups
    geoconv_wmma_kernel<<<grid, 256, 0, stream>>>(
        feat, rho, theta, msk, mur, mut, sr, st, Wc, bc, nrot, out);
}
